// DifferentiableKDEMahalanobis_47966194761999
// MI455X (gfx1250) — matrix operand order repaired and checked by reading, not compiled, not run on hardware
//
#include <hip/hip_runtime.h>
#include <hip/hip_bf16.h>

typedef __attribute__((ext_vector_type(16))) _Float16 v16h;
typedef __attribute__((ext_vector_type(8)))  float    v8f;

#define GRID_W    96               // image is 96x96
#define NPIX      (GRID_W * GRID_W)
#define TDIM      191              // table covers Δ in [-95, 95]
#define TSTRIDE   192              // padded row stride (f16 elements)
#define INV_SQRT_2PI 0.3989422804014327f

// ---------------------------------------------------------------------------
// Kernel 1: build the 191x191 Gaussian stencil table in f16.
// T[d0,d1] = (1/sqrt(2pi)) * exp(-0.5 * [Dx Dy] inv(L L^T) [Dx Dy]^T)
// ---------------------------------------------------------------------------
__global__ void kde_build_table(const float* __restrict__ L,
                                _Float16* __restrict__ T)
{
    // inv of 2x2 SPD cov = L L^T (wave-uniform scalar math)
    float L00 = L[0], L01 = L[1], L10 = L[2], L11 = L[3];
    float c00 = L00 * L00 + L01 * L01;
    float c01 = L00 * L10 + L01 * L11;
    float c11 = L10 * L10 + L11 * L11;
    float invdet = 1.0f / (c00 * c11 - c01 * c01);
    float ia =  c11 * invdet;
    float ib = -c01 * invdet;
    float ic =  c00 * invdet;

    int idx = blockIdx.x * blockDim.x + threadIdx.x;
    if (idx >= TDIM * TDIM) return;
    int d0 = idx / TDIM;
    int d1 = idx - d0 * TDIM;
    float Dx = (float)(d0 - 95);
    float Dy = (float)(d1 - 95);
    float d2 = ia * Dx * Dx + 2.0f * ib * Dx * Dy + ic * Dy * Dy;
    d2 = fmaxf(d2, 0.0f);
    T[d0 * TSTRIDE + d1] = (_Float16)(INV_SQRT_2PI * __expf(-0.5f * d2));
}

// ---------------------------------------------------------------------------
// Kernel 2: full 2D convolution of p (96x96) with the 191x191 stencil,
// expressed as WMMA GEMMs.  One wave owns one 16x16 output tile:
//   O[m,n] += sum_k T[du+95, v_m - jv_k + 95] * p[u_n - du, jv_k]
// accumulated over du in [-95,95] and jv chunks of 32 (K of the f16 WMMA).
// 36 tiles total: 9 blocks x 4 waves.
// ---------------------------------------------------------------------------
__global__ void __launch_bounds__(128)
kde_conv_wmma(const float* __restrict__ samples,   // [4,96,96]; we use the last
              const _Float16* __restrict__ Ttab,   // [191, TSTRIDE] f16
              float* __restrict__ kde_unnorm)      // [9216] f32
{
    __shared__ _Float16 pLds[NPIX];                // 18 KB

    // stage p (last distribution) into LDS as f16
    const float* p = samples + 3 * NPIX;
    for (int i = threadIdx.x; i < NPIX; i += blockDim.x)
        pLds[i] = (_Float16)p[i];
    __syncthreads();

    const int wave = threadIdx.x >> 5;             // 0..3
    const int lane = threadIdx.x & 31;
    const int half = lane >> 4;                    // 0 or 1 (lane group)
    const int l16  = lane & 15;

    const int tile    = blockIdx.x * 4 + wave;     // 0..35
    const int u_base  = (tile % 6) * 16;           // dim-0 (x) tile origin
    const int v_base  = (tile / 6) * 16;           // dim-1 (y) tile origin

    v8f acc = {};                                  // f32 accumulator tile

    for (int du = -95; du <= 95; ++du) {
        const _Float16* Trow = Ttab + (du + 95) * TSTRIDE;
        if (du < 95)                               // hint next stencil row
            __builtin_prefetch(Ttab + (du + 96) * TSTRIDE, 0, 1);

        // B column for this lane: source row jx = u_base + n - du
        const int jx      = u_base + l16 - du;
        const bool jx_ok  = (jx >= 0) && (jx < GRID_W);

        for (int jyb = 0; jyb < GRID_W; jyb += 32) {
            // ---- A operand (16-bit A 16x32, contiguous K per lane half) ------------------
            // element e of lane -> K = e + 16*half, M = l16
            // A[M][K] = Trow[v_base + M - (jyb + K) + 95]
            v16h a;
            const int baseA = v_base + l16 - jyb + 95;   // in [0,190] always
            #pragma unroll
            for (int e = 0; e < 16; ++e) {
                int K = e + half * 16;
                a[e] = Trow[baseA - K];
            }

            // ---- B operand (dense 16-bit B 32x16 layout) ----------------
            // lane half selects K half: element e -> K = e + 16*half, N = l16
            // B[K][N] = p[jx, jyb + K]   (zero when jx out of range)
            v16h b = {};
            if (jx_ok) {
                const _Float16* prow = &pLds[jx * GRID_W + jyb + 16 * half];
                #pragma unroll
                for (int e = 0; e < 16; ++e) b[e] = prow[e];
            }

            // EXEC is full here (divergence confined to the fill above)
            acc = __builtin_amdgcn_wmma_f32_16x16x32_f16(
                      /*neg_a=*/false, a, /*neg_b=*/false, b,
                      /*c_mod=*/(short)0, acc,
                      /*reuse_a=*/false, /*reuse_b=*/false);
        }
    }

    // D layout: VGPR r holds M = r + 8*half, N = l16
    // output flat index = u*96 + v  with u = u_base + N, v = v_base + M
    #pragma unroll
    for (int r = 0; r < 8; ++r) {
        int v = v_base + r + 8 * half;
        int u = u_base + l16;
        kde_unnorm[u * GRID_W + v] = acc[r];
    }
}

// ---------------------------------------------------------------------------
// Kernel 3: normalize  out = kde / sum(kde)   (single block)
// ---------------------------------------------------------------------------
__global__ void __launch_bounds__(256)
kde_normalize(const float* __restrict__ kin, float* __restrict__ out)
{
    __shared__ float red[256];
    float s = 0.0f;
    for (int i = threadIdx.x; i < NPIX; i += 256) s += kin[i];
    red[threadIdx.x] = s;
    __syncthreads();
    for (int off = 128; off > 0; off >>= 1) {
        if ((int)threadIdx.x < off) red[threadIdx.x] += red[threadIdx.x + off];
        __syncthreads();
    }
    float inv = 1.0f / red[0];
    for (int i = threadIdx.x; i < NPIX; i += 256) out[i] = kin[i] * inv;
}

// ---------------------------------------------------------------------------
extern "C" void kernel_launch(void* const* d_in, const int* in_sizes, int n_in,
                              void* d_out, int out_size, void* d_ws, size_t ws_size,
                              hipStream_t stream)
{
    const float* samples = (const float*)d_in[0];   // [4*9216] f32
    const float* L       = (const float*)d_in[1];   // [4] f32
    float* out           = (float*)d_out;           // [9216] f32

    // workspace layout
    _Float16* Ttab = (_Float16*)d_ws;                                   // 191*192*2 = 73344 B
    float* kde_un  = (float*)((char*)d_ws + (size_t)TDIM * TSTRIDE * 2); // 9216*4 B

    // 1) stencil table
    kde_build_table<<<(TDIM * TDIM + 255) / 256, 256, 0, stream>>>(L, Ttab);

    // 2) WMMA convolution: 36 tiles = 9 blocks x 4 waves
    kde_conv_wmma<<<9, 128, 0, stream>>>(samples, Ttab, kde_un);

    // 3) normalize
    kde_normalize<<<1, 256, 0, stream>>>(kde_un, out);
}
